// Gemma4Router_386547057126
// MI455X (gfx1250) — matrix operand order repaired and checked by reading, not compiled, not run on hardware
//
#include <hip/hip_runtime.h>
#include <stdint.h>

// Problem constants (match reference)
#define HD 2048
#define NE 64
#define NK 8
#define NCHUNK (HD / 32)
#define TOK_PER_BLOCK 64
#define NTHREADS 128
#define RSQRT_HD 0.022097086912079608f  // 2048^-0.5
#define NEG_BIG -3.4e38f
#define WSTRIDE 40                      // padded LDS row stride (80B: 16B-aligned, conflict-free)

#if __has_builtin(__builtin_amdgcn_global_load_async_to_lds_b128)
#define USE_ASYNC 1
#else
#define USE_ASYNC 0
#endif

typedef __attribute__((ext_vector_type(16))) __bf16 v16bf;
typedef __attribute__((ext_vector_type(8)))  float  v8f;

#if USE_ASYNC
// Builtin signature (from clang diagnostic): pointer to vector_size(16) int, AS1 src / AS3 dst
typedef int v4i_gcc __attribute__((vector_size(16)));
typedef __attribute__((address_space(1))) v4i_gcc as1_v4i;
typedef __attribute__((address_space(3))) v4i_gcc as3_v4i;
#define TO_AS1(p) ((as1_v4i*)(p))
#define TO_AS3(p) ((as3_v4i*)(p))
#endif

union BF16V {
  v16bf v;
  unsigned int d[8];
  uint4 q[2];
};

__device__ __forceinline__ unsigned int f_as_u(float f) {
  union { float f; unsigned int u; } x; x.f = f; return x.u;
}
__device__ __forceinline__ float u_as_f(unsigned int u) {
  union { unsigned int u; float f; } x; x.u = u; return x.f;
}
// round-to-nearest-even bf16, result in HIGH 16 bits of the returned word
__device__ __forceinline__ unsigned int rne_bf16(unsigned int u) {
  return u + 0x7FFFu + ((u >> 16) & 1u);
}
__device__ __forceinline__ unsigned short f2bf_rn(float f) {
  return (unsigned short)(rne_bf16(f_as_u(f)) >> 16);
}
__device__ __forceinline__ float bf2f(unsigned short h) {
  return u_as_f(((unsigned int)h) << 16);
}

// Fold scale[j] * H^-0.5 into proj_w; split into bf16 hi + bf16 lo (bf16x3 fp32 emulation).
__global__ void prep_weights_kernel(const float* __restrict__ w,
                                    const float* __restrict__ scale,
                                    unsigned short* __restrict__ whi,
                                    unsigned short* __restrict__ wlo) {
  int idx = blockIdx.x * blockDim.x + threadIdx.x;   // < NE*HD
  int j = idx & (HD - 1);
  float wf = w[idx] * scale[j] * RSQRT_HD;
  unsigned short hi = f2bf_rn(wf);
  whi[idx] = hi;
  wlo[idx] = f2bf_rn(wf - bf2f(hi));
}

// Stage one 64x32 hi/lo weight slab into an LDS buffer (each of 128 threads moves 2x16B hi + 2x16B lo)
__device__ __forceinline__ void stage_chunk(const unsigned short* __restrict__ whi,
                                            const unsigned short* __restrict__ wlo,
                                            unsigned short* dhi, unsigned short* dlo,
                                            int c, int tid) {
  const int e = tid >> 1;
  const int part = tid & 1;
  const unsigned short* gh = whi + (size_t)e * HD + c * 32 + part * 16;
  const unsigned short* gl = wlo + (size_t)e * HD + c * 32 + part * 16;
  unsigned short* lh = dhi + e * WSTRIDE + part * 16;
  unsigned short* ll = dlo + e * WSTRIDE + part * 16;
#if USE_ASYNC
  // gfx1250 async copy: global -> LDS without VGPR round trip, tracked by ASYNCcnt
  __builtin_amdgcn_global_load_async_to_lds_b128(TO_AS1(gh),     TO_AS3(lh),     0, 0);
  __builtin_amdgcn_global_load_async_to_lds_b128(TO_AS1(gh + 8), TO_AS3(lh + 8), 0, 0);
  __builtin_amdgcn_global_load_async_to_lds_b128(TO_AS1(gl),     TO_AS3(ll),     0, 0);
  __builtin_amdgcn_global_load_async_to_lds_b128(TO_AS1(gl + 8), TO_AS3(ll + 8), 0, 0);
#else
  uint4 h0 = ((const uint4*)gh)[0];
  uint4 h1 = ((const uint4*)gh)[1];
  uint4 l0 = ((const uint4*)gl)[0];
  uint4 l1 = ((const uint4*)gl)[1];
  *((uint4*)lh)       = h0;
  *((uint4*)(lh + 8)) = h1;
  *((uint4*)ll)       = l0;
  *((uint4*)(ll + 8)) = l1;
#endif
}

__device__ __forceinline__ void wait_async_done() {
#if USE_ASYNC
#if __has_builtin(__builtin_amdgcn_s_wait_asynccnt)
  __builtin_amdgcn_s_wait_asynccnt(0);
#else
  asm volatile("s_wait_asynccnt 0x0" ::: "memory");
#endif
#endif
}

__global__ void __launch_bounds__(NTHREADS)
router_kernel(const float* __restrict__ x,
              const unsigned short* __restrict__ whi,
              const unsigned short* __restrict__ wlo,
              const float* __restrict__ pes,
              float* __restrict__ out_w,
              int* __restrict__ out_i) {
  __shared__ float s_rms[TOK_PER_BLOCK];
  __shared__ unsigned short s_whi[2][NE * WSTRIDE];   // double-buffered weight slabs
  __shared__ unsigned short s_wlo[2][NE * WSTRIDE];
  __shared__ float s_logits[TOK_PER_BLOCK][NE + 2];

  const int tid  = threadIdx.x;
  const int lane = tid & 31;
  const int wave = tid >> 5;
  const int l15  = lane & 15;
  const int kh   = lane >> 4;           // half-wave selects K sub-block per WMMA layout
  const int blockTok = blockIdx.x * TOK_PER_BLOCK;
  const int waveTok  = blockTok + wave * 16;

  // ---- Phase A: per-token sum of squares -> rms scale (coalesced, 512B/instr) ----
  for (int tk = 0; tk < 16; ++tk) {
    const float4* xr = (const float4*)(x + (size_t)(waveTok + tk) * HD);
    float a = 0.f;
    #pragma unroll 4
    for (int i = 0; i < HD / 128; ++i) {          // 16 iters of float4, lane-striped
      float4 v = xr[lane + i * 32];
      a += v.x * v.x + v.y * v.y + v.z * v.z + v.w * v.w;
    }
    #pragma unroll
    for (int off = 16; off > 0; off >>= 1) a += __shfl_xor(a, off, 32);
    if (lane == 0) s_rms[wave * 16 + tk] = rsqrtf(a * (1.0f / HD) + 1e-6f);
  }

  // ---- Phase B: bf16x3 WMMA GEMM, C[16 tokens x 64 experts] per wave ----
  v8f cacc[4] = {};                                 // 4 N-tiles of 16 experts
  const float* arow = x + (size_t)(waveTok + l15) * HD + kh * 8;

  stage_chunk(whi, wlo, s_whi[0], s_wlo[0], 0, tid);   // prologue: stage chunk 0

  for (int c = 0; c < NCHUNK; ++c) {
    const int buf = c & 1;
    wait_async_done();            // this wave's staging of chunk c is in LDS
    __syncthreads();              // ...and everyone else's too
    if (c + 1 < NCHUNK)           // overlap next slab's L2->LDS copy with this chunk's WMMAs
      stage_chunk(whi, wlo, s_whi[buf ^ 1], s_wlo[buf ^ 1], c + 1, tid);

    // A fragment in native 16-bit WMMA A layout:
    // lanes 0-15: row l15, K = c*32 + {0..7, 16..23}; lanes 16-31: K = c*32 + {8..15, 24..31}
    const float* ap = arow + c * 32;
    float4 f0 = ((const float4*)ap)[0];
    float4 f1 = ((const float4*)ap)[1];
    float4 f2 = ((const float4*)(ap + 16))[0];
    float4 f3 = ((const float4*)(ap + 16))[1];
    __builtin_prefetch(ap + 32, 0, 1);              // gfx1250 global_prefetch_b8 for next chunk

    float af[16] = {f0.x, f0.y, f0.z, f0.w, f1.x, f1.y, f1.z, f1.w,
                    f2.x, f2.y, f2.z, f2.w, f3.x, f3.y, f3.z, f3.w};
    BF16V ahi, alo;
    #pragma unroll
    for (int p = 0; p < 8; ++p) {
      // RNE both floats, pack hi pair with one v_perm_b32; exact residual -> RNE lo pair
      unsigned int r0 = rne_bf16(f_as_u(af[2 * p]));
      unsigned int r1 = rne_bf16(f_as_u(af[2 * p + 1]));
      ahi.d[p] = __builtin_amdgcn_perm(r1, r0, 0x07060302u);
      float l0 = af[2 * p]     - u_as_f(r0 & 0xFFFF0000u);   // exact (Dekker split)
      float l1 = af[2 * p + 1] - u_as_f(r1 & 0xFFFF0000u);
      unsigned int t0 = rne_bf16(f_as_u(l0));
      unsigned int t1 = rne_bf16(f_as_u(l1));
      alo.d[p] = __builtin_amdgcn_perm(t1, t0, 0x07060302u);
    }

    #pragma unroll
    for (int n = 0; n < 4; ++n) {
      // B layout: lane = expert column within tile, kh selects K {0..7,16..23} / {8..15,24..31}
      const int off = (n * 16 + l15) * WSTRIDE + kh * 8;
      BF16V bhi, blo;
      bhi.q[0] = *((const uint4*)&s_whi[buf][off]);
      bhi.q[1] = *((const uint4*)&s_whi[buf][off + 16]);
      blo.q[0] = *((const uint4*)&s_wlo[buf][off]);
      blo.q[1] = *((const uint4*)&s_wlo[buf][off + 16]);
      v8f acc = cacc[n];
      acc = __builtin_amdgcn_wmma_f32_16x16x32_bf16(false, ahi.v, false, bhi.v,
                                                    (short)0, acc, false, false);
      acc = __builtin_amdgcn_wmma_f32_16x16x32_bf16(false, ahi.v, false, blo.v,
                                                    (short)0, acc, false, false);
      acc = __builtin_amdgcn_wmma_f32_16x16x32_bf16(false, alo.v, false, bhi.v,
                                                    (short)0, acc, false, false);
      cacc[n] = acc;
    }
  }
  __syncthreads();   // all WMMA reads of the last slab done before the epilogue

  // ---- Epilogue: apply per-token rms scale, scatter logits to LDS ----
  #pragma unroll
  for (int n = 0; n < 4; ++n) {
    #pragma unroll
    for (int v = 0; v < 8; ++v) {
      const int tl = wave * 16 + v + kh * 8;        // C layout: VGPR v -> row, half -> +8
      s_logits[tl][n * 16 + l15] = cacc[n][v] * s_rms[tl];
    }
  }
  __syncthreads();

  // ---- Top-8: softmax 1/sum cancels under renormalization; only 8 exps needed ----
  if (tid < TOK_PER_BLOCK) {
    float* row = s_logits[tid];
    float vals[NK];
    int   idxs[NK];
    #pragma unroll
    for (int k = 0; k < NK; ++k) {
      float m = NEG_BIG;
      int mi = 0;
      for (int e = 0; e < NE; ++e) {
        float v = row[e];
        if (v > m) { m = v; mi = e; }               // strict > keeps lowest index on ties
      }
      vals[k] = m;
      idxs[k] = mi;
      row[mi] = NEG_BIG;
    }
    float mx = vals[0];                             // first argmax is the global max
    float w[NK], s = 0.f;
    #pragma unroll
    for (int k = 0; k < NK; ++k) { w[k] = __expf(vals[k] - mx); s += w[k]; }
    const float inv = 1.0f / s;
    const size_t t = (size_t)(blockTok + tid);
    #pragma unroll
    for (int k = 0; k < NK; ++k) {
      out_w[t * NK + k] = w[k] * inv * pes[idxs[k]];
      out_i[t * NK + k] = idxs[k];
    }
  }
}

extern "C" void kernel_launch(void* const* d_in, const int* in_sizes, int n_in,
                              void* d_out, int out_size, void* d_ws, size_t ws_size,
                              hipStream_t stream) {
  const float* x     = (const float*)d_in[0];   // [T, 2048]
  const float* w     = (const float*)d_in[1];   // [64, 2048]
  const float* scale = (const float*)d_in[2];   // [2048]
  const float* pes   = (const float*)d_in[3];   // [64]
  const int T = in_sizes[0] / HD;

  unsigned short* whi = (unsigned short*)d_ws;            // 64*2048 bf16 (256 KB)
  unsigned short* wlo = whi + (size_t)NE * HD;            // 64*2048 bf16 (256 KB)

  float* out_w = (float*)d_out;                           // [T, 8] f32
  int*   out_i = (int*)d_out + (size_t)T * NK;            // [T, 8] i32 (concatenated)

  prep_weights_kernel<<<(NE * HD) / 256, 256, 0, stream>>>(w, scale, whi, wlo);
  router_kernel<<<T / TOK_PER_BLOCK, NTHREADS, 0, stream>>>(x, whi, wlo, pes, out_w, out_i);
}
